// CTReconModel_54932631715840
// MI455X (gfx1250) — matrix operand order repaired and checked by reading, not compiled, not run on hardware
//
#include <hip/hip_runtime.h>
#include <hip/hip_bf16.h>

typedef __attribute__((ext_vector_type(16))) _Float16 v16h;
typedef __attribute__((ext_vector_type(8)))  _Float16 v8h;
typedef __attribute__((ext_vector_type(8)))  float    v8f;

#define ANGLES 90
#define DET    512
#define IMG    256
#define NNZ    16000000
#define FLT    401
#define ROWS_F (ANGLES*4*DET)   // 184320
#define NPIX   (IMG*IMG)        // 65536

#if defined(__has_builtin)
#if __has_builtin(__builtin_amdgcn_global_load_async_to_lds_b32) && \
    __has_builtin(__builtin_amdgcn_s_wait_asynccnt)
#define USE_ASYNC_LDS 1
#endif
#endif

typedef __attribute__((address_space(1))) int* gint_p;
typedef __attribute__((address_space(3))) int* lint_p;

// Branch-free tanh: 1 - 2/(e^{2x}+1), via v_exp_f32 + v_rcp_f32 (TRANS ops,
// co-executable with WMMA). Activations here are bounded; no special cases.
__device__ __forceinline__ float fast_tanh(float x) {
  float e = __builtin_amdgcn_exp2f(x * 2.8853900817779268f);  // e^{2x}
  return 1.f - 2.f * __builtin_amdgcn_rcpf(e + 1.f);
}

// ---------------------------------------------------------------------------
// prep: build padded sinogram (B,102,524) f16  +  sin_interp (B,90,512,3) f32
// ---------------------------------------------------------------------------
__global__ __launch_bounds__(256) void prep_pad_kernel(const float* __restrict__ sin_in,
                                                       _Float16* __restrict__ pad) {
  int idx = blockIdx.x * 256 + threadIdx.x;
  const int N = 2 * 102 * 524;
  if (idx >= N) return;
  int b  = idx / (102 * 524);
  int r  = idx - b * (102 * 524);
  int ai = r / 524;
  int di = r - ai * 524;
  float v = 0.f;
  if (di >= 6 && di < 518) {
    int a = (ai < 6) ? (84 + ai) : ((ai < 96) ? (ai - 6) : (ai - 96));
    v = sin_in[(b * ANGLES + a) * DET + (di - 6)];
  }
  pad[idx] = (_Float16)v;
}

__global__ __launch_bounds__(256) void prep_interp_kernel(const float* __restrict__ sin_in,
                                                          float* __restrict__ interp) {
  int idx = blockIdx.x * 256 + threadIdx.x;          // over B*90*512
  const int N = 2 * ANGLES * DET;
  if (idx >= N) return;
  int b = idx / (ANGLES * DET);
  int r = idx - b * (ANGLES * DET);
  int a = r / DET;
  int d = r - a * DET;
  int a1 = (a + 1) % ANGLES;
  float s0 = sin_in[(b * ANGLES + a) * DET + d];
  float s1 = sin_in[(b * ANGLES + a1) * DET + d];
#pragma unroll
  for (int c = 0; c < 3; ++c) {
    float coef = (float)(c + 1);
    interp[idx * 3 + c] = ((4.f - coef) * s0 + coef * s1) * 0.25f;
  }
}

// ---------------------------------------------------------------------------
// pack weights into WMMA B-fragment order: [tap][nblock][lane][h] halves.
// B element (k,n): n = nblock*16 + (lane&15), k = (h&7) + (lane>>4)*8 + (h>>3)*16.
// src is (taps, kdim, 32) f32; k >= kdim zero-padded.
// ---------------------------------------------------------------------------
__global__ __launch_bounds__(256) void pack_weights_kernel(const float* __restrict__ src,
                                                           _Float16* __restrict__ dst,
                                                           int taps, int kdim) {
  int idx = blockIdx.x * 256 + threadIdx.x;
  int total = taps * 2 * 32 * 16;
  if (idx >= total) return;
  int h    = idx & 15;
  int lane = (idx >> 4) & 31;
  int nb   = (idx >> 9) & 1;
  int tap  = idx >> 10;
  int n = nb * 16 + (lane & 15);
  int k = (h & 7) + (lane >> 4) * 8 + (h >> 3) * 16;
  float v = (k < kdim) ? src[(tap * kdim + k) * 32 + n] : 0.f;
  dst[idx] = (_Float16)v;
}

// ---------------------------------------------------------------------------
// conv1: 1 -> 32 channels, 5x5 VALID, via one WMMA pair (K=32 = 25 taps + pad)
// in: pad f16 (B,102,524), out: act1 f16 (B,98,520,32) = tanh(conv + b1)
// ---------------------------------------------------------------------------
__global__ __launch_bounds__(256) void conv1_wmma_kernel(const _Float16* __restrict__ pad,
                                                         const _Float16* __restrict__ wpack,
                                                         const float* __restrict__ bias,
                                                         _Float16* __restrict__ out,
                                                         int numTiles) {
  const int OUTH = 98, OUTW = 520, INW = 524, INH = 102;
  const int TOTAL = 2 * OUTH * OUTW;
  int wave = (blockIdx.x * 256 + threadIdx.x) >> 5;
  int lane = threadIdx.x & 31;
  if (wave >= numTiles) return;                    // wave-uniform exit
  int hi = lane >> 4;
  int s0 = wave * 16;
  int s = s0 + (lane & 15);
  if (s >= TOTAL) s = TOTAL - 1;                   // clamp loads, guard stores
  int b = s / (OUTH * OUTW);
  int r = s - b * (OUTH * OUTW);
  int y = r / OUTW;
  int x = r - y * OUTW;
  const _Float16* base = pad + (b * INH + y) * INW + x;

  v16h a;
#pragma unroll
  for (int h = 0; h < 16; ++h) {
    int k = h + ((h < 8) ? (hi ? 8 : 0) : (hi ? 16 : 8));
    int kk = (k < 25) ? k : 0;                     // clamp: unconditional load
    int dy = kk / 5, dx = kk - dy * 5;
    _Float16 v = base[dy * INW + dx];
    if (k >= 25) v = (_Float16)0.f;                // select, not exec-branch
    a[h] = v;
  }
  const v16h* bw = (const v16h*)wpack;
  v16h b0 = bw[lane];
  v16h b1 = bw[32 + lane];
  v8f acc0 = {}, acc1 = {};
  acc0 = __builtin_amdgcn_wmma_f32_16x16x32_f16(false, a, false, b0, (short)0, acc0, false, false);
  acc1 = __builtin_amdgcn_wmma_f32_16x16x32_f16(false, a, false, b1, (short)0, acc1, false, false);

  int n = lane & 15;
  float bn0 = bias[n], bn1 = bias[n + 16];
  if (s0 + 16 <= TOTAL) {                          // uniform full-tile fast path
#pragma unroll
    for (int rr = 0; rr < 8; ++rr) {
      int ss = s0 + rr + hi * 8;
      out[(ss << 5) + n]      = (_Float16)fast_tanh(acc0[rr] + bn0);
      out[(ss << 5) + n + 16] = (_Float16)fast_tanh(acc1[rr] + bn1);
    }
  } else {
#pragma unroll
    for (int rr = 0; rr < 8; ++rr) {
      int ss = s0 + rr + hi * 8;
      if (ss < TOTAL) {
        out[(ss << 5) + n]      = (_Float16)fast_tanh(acc0[rr] + bn0);
        out[(ss << 5) + n + 16] = (_Float16)fast_tanh(acc1[rr] + bn1);
      }
    }
  }
}

// ---------------------------------------------------------------------------
// conv 32->32 5x5 VALID implicit GEMM: 25 taps x (K=32) WMMA, tanh epilogue.
// ---------------------------------------------------------------------------
template <int INH, int INW, int OUTH, int OUTW>
__global__ __launch_bounds__(256) void conv5x5_wmma_kernel(const _Float16* __restrict__ in,
                                                           const _Float16* __restrict__ wpack,
                                                           const float* __restrict__ bias,
                                                           _Float16* __restrict__ out,
                                                           int numTiles) {
  const int TOTAL = 2 * OUTH * OUTW;
  int wave = (blockIdx.x * 256 + threadIdx.x) >> 5;
  int lane = threadIdx.x & 31;
  if (wave >= numTiles) return;                    // wave-uniform exit
  int hi = lane >> 4;
  int s0 = wave * 16;
  int s = s0 + (lane & 15);
  if (s >= TOTAL) s = TOTAL - 1;
  int b = s / (OUTH * OUTW);
  int r = s - b * (OUTH * OUTW);
  int y = r / OUTW;
  int x = r - y * OUTW;
  const _Float16* pix0 = in + (((b * INH + y) * INW + x) << 5);
  int kA0 = hi ? 8 : 0;     // halves 0..7  -> channels kA0..kA0+7
  int kA1 = hi ? 24 : 16;   // halves 8..15 -> channels kA1..kA1+7
  const v16h* bw = (const v16h*)wpack;

  v8f acc0 = {}, acc1 = {};
#pragma unroll
  for (int tap = 0; tap < 25; ++tap) {
    int dy = tap / 5, dx = tap - dy * 5;
    const _Float16* p = pix0 + ((dy * INW + dx) << 5);
    union { v16h v; v8h h[2]; } a;
    a.h[0] = *(const v8h*)(p + kA0);
    a.h[1] = *(const v8h*)(p + kA1);
    v16h bf0 = bw[(tap * 2 + 0) * 32 + lane];
    v16h bf1 = bw[(tap * 2 + 1) * 32 + lane];
    acc0 = __builtin_amdgcn_wmma_f32_16x16x32_f16(false, a.v, false, bf0, (short)0, acc0, false, false);
    acc1 = __builtin_amdgcn_wmma_f32_16x16x32_f16(false, a.v, false, bf1, (short)0, acc1, false, false);
  }

  int n = lane & 15;
  float bn0 = bias[n], bn1 = bias[n + 16];
  if (s0 + 16 <= TOTAL) {                          // uniform full-tile fast path
#pragma unroll
    for (int rr = 0; rr < 8; ++rr) {
      int ss = s0 + rr + hi * 8;
      out[(ss << 5) + n]      = (_Float16)fast_tanh(acc0[rr] + bn0);
      out[(ss << 5) + n + 16] = (_Float16)fast_tanh(acc1[rr] + bn1);
    }
  } else {
#pragma unroll
    for (int rr = 0; rr < 8; ++rr) {
      int ss = s0 + rr + hi * 8;
      if (ss < TOTAL) {
        out[(ss << 5) + n]      = (_Float16)fast_tanh(acc0[rr] + bn0);
        out[(ss << 5) + n + 16] = (_Float16)fast_tanh(acc1[rr] + bn1);
      }
    }
  }
}

// ---------------------------------------------------------------------------
// conv4 (1x1, 32->3) + residual sin_interp, build sin_out and sin_map rows.
// sin_map (B, 360, 512): row a*4   = sin_in, rows a*4+1..3 = sin_out chans.
// ---------------------------------------------------------------------------
__global__ __launch_bounds__(256) void conv4_kernel(const _Float16* __restrict__ act3,
                                                    const float* __restrict__ W4,
                                                    const float* __restrict__ b4,
                                                    const float* __restrict__ interp,
                                                    const float* __restrict__ sin_in,
                                                    float* __restrict__ sin_out,
                                                    float* __restrict__ sin_map) {
  int idx = blockIdx.x * 256 + threadIdx.x;        // over B*90*512
  const int N = 2 * ANGLES * DET;
  if (idx >= N) return;
  int b = idx / (ANGLES * DET);
  int r = idx - b * (ANGLES * DET);
  int a = r / DET;
  int d = r - a * DET;
  const _Float16* p = act3 + ((size_t)idx << 5);
  float h0 = b4[0], h1 = b4[1], h2 = b4[2];
#pragma unroll
  for (int k = 0; k < 32; ++k) {
    float av = (float)p[k];
    h0 = fmaf(av, W4[k * 3 + 0], h0);
    h1 = fmaf(av, W4[k * 3 + 1], h1);
    h2 = fmaf(av, W4[k * 3 + 2], h2);
  }
  float hc[3] = {h0, h1, h2};
  sin_map[((size_t)b * 360 + a * 4) * DET + d] = sin_in[idx];
#pragma unroll
  for (int c = 0; c < 3; ++c) {
    float so = interp[(size_t)idx * 3 + c] + hc[c];
    sin_out[(size_t)idx * 3 + c] = so;
    sin_map[((size_t)b * 360 + a * 4 + 1 + c) * DET + d] = so;
  }
}

// ---------------------------------------------------------------------------
// FBP ramp filter: 401-tap SAME cross-correlation along detector, per row.
// One workgroup per row; row body staged to LDS with CDNA5 async-to-LDS when
// the toolchain exposes it, else direct loads.
// ---------------------------------------------------------------------------
__global__ __launch_bounds__(256) void fbp_filter_kernel(const float* __restrict__ sin_map,
                                                         const float* __restrict__ flt,
                                                         float* __restrict__ sin_flat) {
  __shared__ float rowpad[DET + 2 * 200];          // 912
  __shared__ float kbuf[FLT];
  int row = blockIdx.x;                            // 0..719
  const float* src = sin_map + (size_t)row * DET;
#if defined(USE_ASYNC_LDS)
  {
    int i = threadIdx.x;
    if (i < 200) { rowpad[i] = 0.f; rowpad[712 + i] = 0.f; }  // zero halos
    __builtin_amdgcn_global_load_async_to_lds_b32(
        (gint_p)(src + i),       (lint_p)(rowpad + 200 + i), 0, 0);
    __builtin_amdgcn_global_load_async_to_lds_b32(
        (gint_p)(src + 256 + i), (lint_p)(rowpad + 456 + i), 0, 0);
    for (int j = i; j < FLT; j += 256) kbuf[j] = flt[j];
    __builtin_amdgcn_s_wait_asynccnt(0);
  }
#else
  for (int i = threadIdx.x; i < DET + 400; i += 256) {
    int j = i - 200;
    rowpad[i] = (j >= 0 && j < DET) ? src[j] : 0.f;
  }
  for (int i = threadIdx.x; i < FLT; i += 256) kbuf[i] = flt[i];
#endif
  __syncthreads();
  for (int x = threadIdx.x; x < DET; x += 256) {
    float acc = 0.f;
#pragma unroll 4
    for (int i = 0; i < FLT; ++i) acc = fmaf(kbuf[i], rowpad[x + i], acc);
    sin_flat[(size_t)row * DET + x] = acc;
  }
}

// ---------------------------------------------------------------------------
// sparse back-projection: stream 16M (val,row,col) as b128 packets (4 NNZ per
// iteration), atomically scatter both batches (accumulator is L2-resident).
// ---------------------------------------------------------------------------
__global__ __launch_bounds__(256) void sparse_kernel(const float* __restrict__ vals,
                                                     const int* __restrict__ rows,
                                                     const int* __restrict__ cols,
                                                     const float* __restrict__ sin_flat,
                                                     float* __restrict__ acc) {
  const int NQ = NNZ / 4;                          // 16M divisible by 4
  int i = blockIdx.x * 256 + threadIdx.x;
  int stride = gridDim.x * 256;
  const float4* v4 = (const float4*)vals;
  const int4*   r4 = (const int4*)rows;
  const int4*   c4 = (const int4*)cols;
  for (; i < NQ; i += stride) {
    if (i + stride < NQ) {
      __builtin_prefetch(v4 + i + stride, 0, 1);   // global_prefetch_b8
      __builtin_prefetch(r4 + i + stride, 0, 1);
      __builtin_prefetch(c4 + i + stride, 0, 1);
    }
    float4 v = v4[i];
    int4   r = r4[i];
    int4   c = c4[i];
    atomicAdd(&acc[c.x],        sin_flat[r.x] * v.x);
    atomicAdd(&acc[NPIX + c.x], sin_flat[ROWS_F + r.x] * v.x);
    atomicAdd(&acc[c.y],        sin_flat[r.y] * v.y);
    atomicAdd(&acc[NPIX + c.y], sin_flat[ROWS_F + r.y] * v.y);
    atomicAdd(&acc[c.z],        sin_flat[r.z] * v.z);
    atomicAdd(&acc[NPIX + c.z], sin_flat[ROWS_F + r.z] * v.z);
    atomicAdd(&acc[c.w],        sin_flat[r.w] * v.w);
    atomicAdd(&acc[NPIX + c.w], sin_flat[ROWS_F + r.w] * v.w);
  }
}

__global__ __launch_bounds__(256) void zero_kernel(float* __restrict__ p, int n) {
  int i = blockIdx.x * 256 + threadIdx.x;
  if (i < n) p[i] = 0.f;
}

__global__ __launch_bounds__(256) void finalize_kernel(const float* __restrict__ acc,
                                                       const float* __restrict__ scale,
                                                       const float* __restrict__ bias,
                                                       float* __restrict__ fbp_out,
                                                       float* __restrict__ ct_out, int n) {
  int i = blockIdx.x * 256 + threadIdx.x;
  if (i < n) {
    float v = acc[i] * scale[0] + bias[0];
    fbp_out[i] = v;
    ct_out[i]  = v;
  }
}

// ---------------------------------------------------------------------------
extern "C" void kernel_launch(void* const* d_in, const int* in_sizes, int n_in,
                              void* d_out, int out_size, void* d_ws, size_t ws_size,
                              hipStream_t stream) {
  (void)in_sizes; (void)n_in; (void)out_size; (void)ws_size;
  const float* sin_in = (const float*)d_in[0];
  const float* W1     = (const float*)d_in[1];
  const float* b1     = (const float*)d_in[2];
  const float* W2     = (const float*)d_in[3];
  const float* b2     = (const float*)d_in[4];
  const float* W3     = (const float*)d_in[5];
  const float* b3     = (const float*)d_in[6];
  const float* W4     = (const float*)d_in[7];
  const float* b4     = (const float*)d_in[8];
  const float* flt    = (const float*)d_in[9];
  const float* scale  = (const float*)d_in[10];
  const float* fbias  = (const float*)d_in[11];
  const float* A_vals = (const float*)d_in[12];
  const int*   A_rows = (const int*)d_in[13];
  const int*   A_cols = (const int*)d_in[14];

  float* out     = (float*)d_out;                 // all-f32 tuple, flat concat
  float* sin_out = out;                           // (2,90,512,3)  = 276480
  float* fbp_out = out + 276480;                  // (2,256,256,1) = 131072
  float* ct_out  = out + 276480 + 131072;         // same           131072
  float* interp  = out + 276480 + 2 * 131072;     // (2,90,512,3)  = 276480

  char* ws = (char*)d_ws;
  size_t off = 0;
  auto alloc = [&](size_t bytes) -> void* {
    void* p = ws + off;
    off = (off + bytes + 255) & ~(size_t)255;
    return p;
  };
  _Float16* pad    = (_Float16*)alloc((size_t)2 * 102 * 524 * 2);
  _Float16* act1   = (_Float16*)alloc((size_t)2 * 98 * 520 * 32 * 2);
  _Float16* act2   = (_Float16*)alloc((size_t)2 * 94 * 516 * 32 * 2);
  _Float16* act3   = (_Float16*)alloc((size_t)2 * 90 * 512 * 32 * 2);
  _Float16* w1pack = (_Float16*)alloc((size_t)1 * 2 * 32 * 16 * 2);
  _Float16* w2pack = (_Float16*)alloc((size_t)25 * 2 * 32 * 16 * 2);
  _Float16* w3pack = (_Float16*)alloc((size_t)25 * 2 * 32 * 16 * 2);
  float* sin_map   = (float*)alloc((size_t)2 * 360 * 512 * 4);
  float* sin_flat  = (float*)alloc((size_t)2 * 360 * 512 * 4);
  float* fbp_acc   = (float*)alloc((size_t)2 * NPIX * 4);

  // stage 0: zero accumulator (graph-replay safe: re-zeroed every call)
  zero_kernel<<<(2 * NPIX + 255) / 256, 256, 0, stream>>>(fbp_acc, 2 * NPIX);

  // stage 1: padding + interpolation
  prep_pad_kernel<<<(2 * 102 * 524 + 255) / 256, 256, 0, stream>>>(sin_in, pad);
  prep_interp_kernel<<<(2 * ANGLES * DET + 255) / 256, 256, 0, stream>>>(sin_in, interp);

  // stage 2: pack conv weights into WMMA B-fragment order
  pack_weights_kernel<<<(1024 + 255) / 256, 256, 0, stream>>>(W1, w1pack, 1, 25);
  pack_weights_kernel<<<(25600 + 255) / 256, 256, 0, stream>>>(W2, w2pack, 25, 32);
  pack_weights_kernel<<<(25600 + 255) / 256, 256, 0, stream>>>(W3, w3pack, 25, 32);

  // stage 3: conv stack (WMMA implicit GEMM, f16 in / f32 acc / f16 out)
  {
    int tiles = (2 * 98 * 520 + 15) / 16;
    conv1_wmma_kernel<<<(tiles + 7) / 8, 256, 0, stream>>>(pad, w1pack, b1, act1, tiles);
  }
  {
    int tiles = (2 * 94 * 516 + 15) / 16;
    conv5x5_wmma_kernel<98, 520, 94, 516><<<(tiles + 7) / 8, 256, 0, stream>>>(act1, w2pack, b2, act2, tiles);
  }
  {
    int tiles = (2 * 90 * 512 + 15) / 16;
    conv5x5_wmma_kernel<94, 516, 90, 512><<<(tiles + 7) / 8, 256, 0, stream>>>(act2, w3pack, b3, act3, tiles);
  }

  // stage 4: 1x1 conv + residual, assemble interleaved sinogram map
  conv4_kernel<<<(2 * ANGLES * DET + 255) / 256, 256, 0, stream>>>(act3, W4, b4, interp, sin_in,
                                                                   sin_out, sin_map);

  // stage 5: FBP ramp filter (one workgroup per sinogram row, LDS staging)
  fbp_filter_kernel<<<2 * 360, 256, 0, stream>>>(sin_map, flt, sin_flat);

  // stage 6: sparse back-projection scatter (bandwidth-bound stream)
  sparse_kernel<<<4096, 256, 0, stream>>>(A_vals, A_rows, A_cols, sin_flat, fbp_acc);

  // stage 7: scale + bias into fbp_out and ct_out
  finalize_kernel<<<(2 * NPIX + 255) / 256, 256, 0, stream>>>(fbp_acc, scale, fbias,
                                                              fbp_out, ct_out, 2 * NPIX);
}
